// BiLSTM_CRF_43593918054723
// MI455X (gfx1250) — compile-verified, matrix operand order checked by reading, not run on hardware
//
#include <hip/hip_runtime.h>
#include <hip/hip_bf16.h>
#include <stddef.h>

// Problem dims (fixed by the reference)
#define B_ 64
#define T_ 256
#define E_ 256
#define HD_ 256
#define G4_ 1024           // 4*Hd
#define M_ (B_ * T_)       // 16384 token rows, row index m = t*B + b
#define KTAG 12
#define START_TAG 10

typedef _Float16 v16h __attribute__((ext_vector_type(16)));
typedef _Float16 v8h  __attribute__((ext_vector_type(8)));
typedef float    v8f  __attribute__((ext_vector_type(8)));
typedef unsigned int v4u __attribute__((ext_vector_type(4)));
typedef int      v8i  __attribute__((ext_vector_type(8)));
typedef int      v4i  __attribute__((ext_vector_type(4)));

// Codegen-confirmed CDNA5 WMMA: D(f32 16x16) = A(f16 16x32) x B(f16 32x16) + C
__device__ __forceinline__ v8f wmma16(v16h a, v16h b, v8f c) {
  return __builtin_amdgcn_wmma_f32_16x16x32_f16(false, a, false, b, (short)0, c,
                                                false, false);
}

// K-index inside a 16x32 f16 A-fragment (or 32x16 B-fragment, lane<->N) for
// (lane, element i of v16h), per ISA 7.12.2 "16-bit A-Matrix 16x32" layout.
__device__ __forceinline__ int frag_k(int lane, int i) {
  int v = i >> 1, h = i & 1;
  if (lane < 16) return (v < 4) ? (2 * v + h) : (16 + 2 * (v - 4) + h);
  return (v < 4) ? (8 + 2 * v + h) : (24 + 2 * (v - 4) + h);
}

// Load a 16x32 A-fragment from a row-major f16 matrix (global or LDS):
// per lane: two contiguous 16B chunks -> 2x b128 loads.
__device__ __forceinline__ v16h load_a_rm(const _Float16* X, int ld, int m0,
                                          int k0, int lane) {
  int r = lane & 15;
  int koff = (lane & 16) ? 8 : 0;
  const _Float16* p = X + (size_t)(m0 + r) * ld + k0 + koff;
  v8h lo = *(const v8h*)p;
  v8h hi = *(const v8h*)(p + 16);
  v16h a;
#pragma unroll
  for (int i = 0; i < 8; ++i) { a[i] = lo[i]; a[8 + i] = hi[i]; }
  return a;
}

__device__ __forceinline__ float sigf(float x) { return 1.0f / (1.0f + __expf(-x)); }

// ---------------------------------------------------------------------------
// Tensor Data Mover: DMA a 2D tile (1024 f16 x 64 rows, row stride 2048 f16)
// from global memory into LDS at byte offset lds_off. D# packed per ISA
// "Tensor DMA Descriptor" (group0: count=1, type=2; group1: data_size=2B,
// dims/strides). Issued once per executing wave; tracked with TENSORcnt.
// This toolchain exposes the 6-arg builtin (g0,g1,g2,g3,g4,cpol).
// ---------------------------------------------------------------------------
__device__ __forceinline__ void tdm_load_slab(const void* gsrc, unsigned lds_off) {
  unsigned long long ga = (unsigned long long)gsrc;
  v4u g0 = { 1u,                                   // count=1 (user mode)
             lds_off,                              // lds_addr (bytes)
             (unsigned)(ga & 0xFFFFFFFFu),         // global_addr[31:0]
             (unsigned)((ga >> 32) & 0x1FFFFFFull) // global_addr[56:32]
               | 0x80000000u };                    // type=2 -> bits[127:126]=10b
  v8i g1 = { (int)(1u << 16),    // wg_mask=0, data_size=1 (2 bytes)
             (int)(1024u << 16), // tensor_dim0[15:0]=1024  (bits 63:48)
             (int)(64u << 16),   // tensor_dim0[31:16]=0, tensor_dim1[15:0]=64
             (int)(1024u << 16), // tensor_dim1[31:16]=0, tile_dim0=1024
             (int)64,            // tile_dim1=64, tile_dim2=0
             (int)2048,          // tensor_dim0_stride[31:0]=2048 elements
             0, 0 };             // stride hi / tensor_dim1_stride = 0
  v4i z4 = { 0, 0, 0, 0 };
  v8i z8 = { 0, 0, 0, 0, 0, 0, 0, 0 };
  __builtin_amdgcn_tensor_load_to_lds(g0, g1, z4, z4, z8, 0);
}

// ---------------------------------------------------------------------------
// Prep: swizzle all weights into WMMA B-operand lane order (f16), fold biases.
// ---------------------------------------------------------------------------
__global__ void k_prep(const float* __restrict__ Wih_f, const float* __restrict__ Wih_b,
                       const float* __restrict__ Whh_f, const float* __restrict__ Whh_b,
                       const float* __restrict__ bih_f, const float* __restrict__ bhh_f,
                       const float* __restrict__ bih_b, const float* __restrict__ bhh_b,
                       const float* __restrict__ Wp,
                       _Float16* __restrict__ Wsw_in, _Float16* __restrict__ Whh_sw,
                       _Float16* __restrict__ Wp_sw, float* __restrict__ biasc) {
  int id = blockIdx.x * blockDim.x + threadIdx.x;
  if (id < 524288) {  // input weights: frag = kslab*128 + ntile
    int i = id & 15, lane = (id >> 4) & 31, frag = id >> 9;
    int ntile = frag & 127, kslab = frag >> 7;
    int kk = kslab * 32 + frag_k(lane, i);
    int n = ntile * 16 + (lane & 15);
    float w = (n < G4_) ? Wih_f[n * E_ + kk] : Wih_b[(n - G4_) * E_ + kk];
    Wsw_in[id] = (_Float16)w;
    return;
  }
  id -= 524288;
  if (id < 524288) {  // recurrent weights, dir-major: frag = kslab*64 + ntile
    int dir = id >> 18, e = id & 262143;
    int i = e & 15, lane = (e >> 4) & 31, frag = e >> 9;
    int ntile = frag & 63, kslab = frag >> 6;
    int kk = kslab * 32 + frag_k(lane, i);
    int n = ntile * 16 + (lane & 15);
    const float* W = dir ? Whh_b : Whh_f;
    Whh_sw[id] = (_Float16)W[n * HD_ + kk];
    return;
  }
  id -= 524288;
  if (id < 8192) {  // projection weights: kslab = id>>9, single ntile
    int i = id & 15, lane = (id >> 4) & 31, kslab = id >> 9;
    int kk = kslab * 32 + frag_k(lane, i);
    int n = lane & 15;
    Wp_sw[id] = (_Float16)((n < KTAG) ? Wp[n * 512 + kk] : 0.0f);
    return;
  }
  id -= 8192;
  if (id < 2048) {
    biasc[id] = (id < G4_) ? (bih_f[id] + bhh_f[id])
                           : (bih_b[id - G4_] + bhh_b[id - G4_]);
  }
}

// ---------------------------------------------------------------------------
// Embedding gather -> f16, time-major rows m = t*B + b.
// ---------------------------------------------------------------------------
__global__ void k_gather(const int* __restrict__ sentence,
                         const float* __restrict__ emb,
                         _Float16* __restrict__ x_h) {
  int m = blockIdx.x, e = threadIdx.x;
  int t = m / B_, b = m % B_;
  int tok = sentence[b * T_ + t];
  x_h[(size_t)m * E_ + e] = (_Float16)emb[(size_t)tok * E_ + e];
}

// ---------------------------------------------------------------------------
// Input GEMM: Gx[M_,2048](f16) = x_h[M_,E_] @ [Wih_f.T|Wih_b.T] + (bih+bhh).
// Each wave computes 4 adjacent N-tiles sharing one A-fragment stream:
// per K-slab: 1 A load + 4 B loads -> 4 WMMAs.
// ---------------------------------------------------------------------------
__global__ void k_ingemm(const _Float16* __restrict__ x_h,
                         const _Float16* __restrict__ Wsw,
                         const float* __restrict__ biasc,
                         _Float16* __restrict__ Gx) {
  int wave = threadIdx.x >> 5, lane = threadIdx.x & 31;
  int task = blockIdx.x * 8 + wave;          // 32768 tasks
  int ng = task & 31, mtile = task >> 5;     // ntile group of 4
  int m0 = mtile * 16;
  int col = lane & 15, rowb = (lane & 16) ? 8 : 0;
  v8f acc[4];
#pragma unroll
  for (int j = 0; j < 4; ++j) {
    float bias = biasc[(ng * 4 + j) * 16 + col];
#pragma unroll
    for (int r = 0; r < 8; ++r) acc[j][r] = bias;
  }
#pragma unroll
  for (int ks = 0; ks < 8; ++ks) {
    v16h a = load_a_rm(x_h, E_, m0, ks * 32, lane);
#pragma unroll
    for (int j = 0; j < 4; ++j) {
      int ntile = ng * 4 + j;
      v16h bf = *(const v16h*)(Wsw + ((size_t)(ks * 128 + ntile) * 32 + lane) * 16);
      acc[j] = wmma16(a, bf, acc[j]);
    }
  }
#pragma unroll
  for (int j = 0; j < 4; ++j)
#pragma unroll
    for (int r = 0; r < 8; ++r)
      Gx[(size_t)(m0 + rowb + r) * 2048 + (ng * 4 + j) * 16 + col] =
          (_Float16)acc[j][r];
}

// ---------------------------------------------------------------------------
// LSTM scan. One workgroup per direction, 512 threads = 16 waves.
// LDS (288 KB dynamic): h(f16, 32 KB) + double-buffered Gx slab (2x128 KB).
// The Tensor Data Mover prefetches step t+1's gate slab while step t's
// recurrent WMMAs run; cell state c lives in registers.
// ---------------------------------------------------------------------------
__global__ void __launch_bounds__(512)
k_lstm(const _Float16* __restrict__ Gx, const _Float16* __restrict__ Whh_sw,
       const float* __restrict__ h0, const float* __restrict__ c0,
       _Float16* __restrict__ hs) {
  extern __shared__ char smem[];
  _Float16* sh_h  = (_Float16*)smem;            // [64*256]   offset 0
  _Float16* sh_gx = (_Float16*)(smem + 32768);  // [2][64*1024]
  const unsigned GX_LDS_OFF = 32768u;

  const int dir = blockIdx.x;
  const int tid = threadIdx.x, wave = tid >> 5, lane = tid & 31;
  const int col = lane & 15, rowb = (lane & 16) ? 8 : 0;
  const _Float16* Wd = Whh_sw + (size_t)dir * HD_ * G4_;

  // Kick off the TDM for step 0's slab immediately.
  if (wave == 0) {
    int tt0 = dir ? (T_ - 1) : 0;
    tdm_load_slab(Gx + (size_t)(tt0 * B_) * 2048 + dir * G4_, GX_LDS_OFF);
  }

  // Warm the recurrent weights into cache (global_prefetch_b8).
  for (size_t p = (size_t)tid * 256; p < (size_t)HD_ * G4_ * 2; p += 512u * 256u)
    __builtin_prefetch((const char*)Wd + p, 0, 1);

  for (int idx = tid; idx < B_ * HD_; idx += 512)
    sh_h[idx] = (_Float16)h0[dir * B_ * HD_ + idx];

  float creg[4][8];
#pragma unroll
  for (int q = 0; q < 4; ++q) {
    int grp = wave * 4 + q, mtile = grp >> 4, nh = grp & 15;
#pragma unroll
    for (int r = 0; r < 8; ++r) {
      int m = mtile * 16 + rowb + r, hc = nh * 16 + col;
      creg[q][r] = c0[dir * B_ * HD_ + m * HD_ + hc];
    }
  }
  if (wave == 0) __builtin_amdgcn_s_wait_tensorcnt(0);
  __syncthreads();

  for (int step = 0; step < T_; ++step) {
    const int tt = dir ? (T_ - 1 - step) : step;
    // Prefetch next step's slab into the other buffer while we compute.
    if (wave == 0 && step + 1 < T_) {
      int tn = dir ? (T_ - 2 - step) : (step + 1);
      tdm_load_slab(Gx + (size_t)(tn * B_) * 2048 + dir * G4_,
                    GX_LDS_OFF + (unsigned)((step + 1) & 1) * 131072u);
    }
    const _Float16* slab = sh_gx + (size_t)(step & 1) * 65536;

    float hnew[4][8];
#pragma unroll
    for (int q = 0; q < 4; ++q) {
      int grp = wave * 4 + q, mtile = grp >> 4, nh = grp & 15;
      v8f acc[4];
#pragma unroll
      for (int g = 0; g < 4; ++g)
#pragma unroll
        for (int r = 0; r < 8; ++r)
          acc[g][r] = (float)slab[(mtile * 16 + rowb + r) * G4_ +
                                  g * HD_ + nh * 16 + col];
#pragma unroll
      for (int ks = 0; ks < 8; ++ks) {
        v16h a = load_a_rm(sh_h, HD_, mtile * 16, ks * 32, lane);  // ds_load
#pragma unroll
        for (int g = 0; g < 4; ++g) {
          int ntile = g * 16 + nh;
          v16h bf = *(const v16h*)(Wd + ((size_t)(ks * 64 + ntile) * 32 + lane) * 16);
          acc[g] = wmma16(a, bf, acc[g]);
        }
      }
      // gate order i,f,g,o (PyTorch)
#pragma unroll
      for (int r = 0; r < 8; ++r) {
        float iv = sigf(acc[0][r]);
        float fv = sigf(acc[1][r]);
        float gv = tanhf(acc[2][r]);
        float ov = sigf(acc[3][r]);
        float cn = fv * creg[q][r] + iv * gv;
        creg[q][r] = cn;
        hnew[q][r] = ov * tanhf(cn);
      }
    }
    __syncthreads();  // everyone finished reading sh_h for this step
#pragma unroll
    for (int q = 0; q < 4; ++q) {
      int grp = wave * 4 + q, mtile = grp >> 4, nh = grp & 15;
#pragma unroll
      for (int r = 0; r < 8; ++r) {
        int m = mtile * 16 + rowb + r, hc = nh * 16 + col;
        _Float16 hh = (_Float16)hnew[q][r];
        sh_h[m * HD_ + hc] = hh;
        hs[((size_t)dir * M_ + (size_t)tt * B_ + m) * HD_ + hc] = hh;
      }
    }
    // Next step's slab must be in LDS before anyone reads it.
    if (wave == 0) __builtin_amdgcn_s_wait_tensorcnt(0);
    __syncthreads();
  }
}

// ---------------------------------------------------------------------------
// Projection: feats[M_,16] = [hf|hb] @ Wp.T + bp (cols >=12 are dummies).
// ---------------------------------------------------------------------------
__global__ void k_proj(const _Float16* __restrict__ hs,
                       const _Float16* __restrict__ Wp_sw,
                       const float* __restrict__ bp,
                       float* __restrict__ feats) {
  int wave = threadIdx.x >> 5, lane = threadIdx.x & 31;
  int mtile = blockIdx.x * 4 + wave;   // 1024 tiles
  int m0 = mtile * 16;
  int col = lane & 15, rowb = (lane & 16) ? 8 : 0;
  float bias = (col < KTAG) ? bp[col] : -1.0e30f;
  v8f acc;
#pragma unroll
  for (int r = 0; r < 8; ++r) acc[r] = bias;
#pragma unroll
  for (int ks = 0; ks < 16; ++ks) {
    const _Float16* src = hs + (size_t)(ks >> 3) * M_ * HD_;  // hf then hb
    v16h a = load_a_rm(src, HD_, m0, (ks & 7) * 32, lane);
    v16h bf = *(const v16h*)(Wp_sw + ((size_t)ks * 32 + lane) * 16);
    acc = wmma16(a, bf, acc);
  }
#pragma unroll
  for (int r = 0; r < 8; ++r)
    feats[(size_t)(m0 + rowb + r) * 16 + col] = acc[r];
}

// ---------------------------------------------------------------------------
// Viterbi: one wave32 per batch row, lane i = tag i; score broadcast via
// wave-synchronous shuffles; backtrack on lane 0.
// ---------------------------------------------------------------------------
__global__ void k_viterbi(const int* __restrict__ sentence,
                          const float* __restrict__ feats,
                          const float* __restrict__ trans,
                          int* __restrict__ bptr,
                          float* __restrict__ out_paths,
                          float* __restrict__ out_score) {
  int b = blockIdx.x, lane = threadIdx.x;

  int cnt = 0;
  for (int t = lane; t < T_; t += 32) cnt += (sentence[b * T_ + t] > 0) ? 1 : 0;
#pragma unroll
  for (int off = 16; off > 0; off >>= 1) cnt += __shfl_down(cnt, off, 32);
  int len = __shfl(cnt, 0, 32);

  float trow[KTAG];
#pragma unroll
  for (int j = 0; j < KTAG; ++j)
    trow[j] = (lane < KTAG) ? trans[lane * KTAG + j] : 0.0f;

  float score = (lane == START_TAG) ? 0.0f : -10000.0f;
  if (lane >= KTAG) score = -3.0e38f;

  for (int t = 0; t < T_; ++t) {
    float best = -3.0e38f;
    int barg = 0;
#pragma unroll
    for (int j = 0; j < KTAG; ++j) {
      float sj = __shfl(score, j, 32);
      float cand = sj + trow[j];
      if (cand > best) { best = cand; barg = j; }
    }
    float ns = best + ((lane < KTAG) ? feats[((size_t)t * B_ + b) * 16 + lane] : 0.0f);
    if (lane < KTAG) bptr[((size_t)t * B_ + b) * KTAG + lane] = barg;
    score = (lane < KTAG) ? ns : -3.0e38f;
  }

  float m = -3.0e38f;
  int a = 0;
#pragma unroll
  for (int j = 0; j < KTAG; ++j) {
    float sj = __shfl(score, j, 32);
    if (sj > m) { m = sj; a = j; }
  }
  if (lane == 0) {
    out_score[b] = m;
    int x = a;
    for (int t = T_ - 1; t >= 0; --t) {
      int emit = x;
      int nxt = bptr[((size_t)t * B_ + b) * KTAG + x];
      if (t < len) x = nxt;
      out_paths[b * T_ + t] = (t < len) ? (float)emit : -1.0f;
    }
  }
}

// ---------------------------------------------------------------------------
extern "C" void kernel_launch(void* const* d_in, const int* in_sizes, int n_in,
                              void* d_out, int out_size, void* d_ws, size_t ws_size,
                              hipStream_t stream) {
  (void)in_sizes; (void)n_in; (void)out_size; (void)ws_size;
  const int*   sentence = (const int*)d_in[0];
  const float* emb      = (const float*)d_in[2];
  const float* Wih_f    = (const float*)d_in[3];
  const float* Whh_f    = (const float*)d_in[4];
  const float* bih_f    = (const float*)d_in[5];
  const float* bhh_f    = (const float*)d_in[6];
  const float* Wih_b    = (const float*)d_in[7];
  const float* Whh_b    = (const float*)d_in[8];
  const float* bih_b    = (const float*)d_in[9];
  const float* bhh_b    = (const float*)d_in[10];
  const float* Wp       = (const float*)d_in[11];
  const float* bp       = (const float*)d_in[12];
  const float* trans    = (const float*)d_in[13];
  const float* h0       = (const float*)d_in[14];
  const float* c0       = (const float*)d_in[15];

  char* ws = (char*)d_ws;
  size_t off = 0;
  auto take = [&](size_t bytes) -> void* {
    void* p = (void*)(ws + off);
    off = (off + bytes + 255) & ~(size_t)255;
    return p;
  };
  _Float16* x_h    = (_Float16*)take((size_t)M_ * E_ * 2);        //   8 MB
  _Float16* Wsw_in = (_Float16*)take((size_t)E_ * 2048 * 2);      //   1 MB
  _Float16* Whh_sw = (_Float16*)take((size_t)2 * HD_ * G4_ * 2);  //   1 MB
  _Float16* Wp_sw  = (_Float16*)take((size_t)512 * 16 * 2);       //  16 KB
  float*    biasc  = (float*)take((size_t)2048 * 4);              //   8 KB
  _Float16* Gx     = (_Float16*)take((size_t)M_ * 2048 * 2);      //  67 MB
  _Float16* hs     = (_Float16*)take((size_t)2 * M_ * HD_ * 2);   //  16 MB
  float*    feats  = (float*)take((size_t)M_ * 16 * 4);           //   1 MB
  int*      bptr   = (int*)take((size_t)T_ * B_ * KTAG * 4);      // 768 KB

  k_prep<<<4136, 256, 0, stream>>>(Wih_f, Wih_b, Whh_f, Whh_b, bih_f, bhh_f,
                                   bih_b, bhh_b, Wp, Wsw_in, Whh_sw, Wp_sw, biasc);
  k_gather<<<M_, 256, 0, stream>>>(sentence, emb, x_h);
  k_ingemm<<<4096, 256, 0, stream>>>(x_h, Wsw_in, biasc, Gx);
  k_lstm<<<2, 512, 294912, stream>>>(Gx, Whh_sw, h0, c0, hs);
  k_proj<<<256, 128, 0, stream>>>(hs, Wp_sw, bp, feats);

  float* out_paths = (float*)d_out;
  float* out_score = out_paths + (size_t)B_ * T_;
  k_viterbi<<<B_, 32, 0, stream>>>(sentence, feats, trans, bptr, out_paths, out_score);
}
